// ConvAttnLSTM_45320494907806
// MI455X (gfx1250) — matrix operand order repaired and checked by reading, not compiled, not run on hardware
//
#include <hip/hip_runtime.h>
#include <math.h>

typedef __attribute__((ext_vector_type(16))) _Float16 v16h;
typedef __attribute__((ext_vector_type(8)))  _Float16 v8h;
typedef __attribute__((ext_vector_type(8)))  float    v8f;

#define LAYERS 2
#define BATCH  128
#define CIN    64
#define EDIM   64
#define HGT    8
#define WID    8
#define NHEAD  8
#define MEMN   40
#define HW     64          // HGT*WID
#define THD    512         // (EDIM/NHEAD)*HW
#define MROWS  (BATCH*HW)  // 8192 GEMM rows
#define NEGB   -1e9f
#define MASKB  3.0f

#define MT 4               // M-tiles (16 rows each) per wave = one batch sample
#define NT 2               // N-tiles (16 cols each) per wave

// ---------------------------------------------------------------------------
// bias[b][m]: rolling attention mask. m<39 uses src_mask[b][m+1] | !notdone[b];
// slot 39 is forced open with MASKB.
__global__ void bias_kernel(const unsigned char* __restrict__ src_mask,
                            const unsigned char* __restrict__ notdone,
                            float* __restrict__ bias) {
  int idx = blockIdx.x * 256 + threadIdx.x;
  if (idx >= BATCH * MEMN) return;
  int b = idx / MEMN, m = idx - b * MEMN;
  float v;
  if (m == MEMN - 1) v = MASKB;
  else v = (src_mask[b * MEMN + m + 1] || !notdone[b]) ? NEGB : 0.0f;
  bias[idx] = v;
}

// out_cur = h0[L-1] * notdone  (NCHW f32)
__global__ void init_out_kernel(const float* __restrict__ h_last,
                                const unsigned char* __restrict__ notdone,
                                float* __restrict__ outc) {
  int idx = blockIdx.x * 256 + threadIdx.x;
  if (idx >= BATCH * EDIM * HW) return;
  int b = idx >> 12;  // / 4096
  outc[idx] = h_last[idx] * (notdone[b] ? 1.0f : 0.0f);
}

// in192 f16 NHWC [b][hw][192] = concat(x, out_cur, h0[n]*nd) converted to f16
__global__ void build_input_kernel(const float* __restrict__ x,
                                   const float* __restrict__ outc,
                                   const float* __restrict__ h,
                                   const unsigned char* __restrict__ notdone,
                                   _Float16* __restrict__ in192) {
  int idx = blockIdx.x * 256 + threadIdx.x;
  if (idx >= BATCH * HW * 192) return;
  int c = idx % 192;
  int rest = idx / 192;
  int hw = rest & 63, b = rest >> 6;
  float v;
  if (c < 64)       v = x[((size_t)b * 64 + c) * 64 + hw];
  else if (c < 128) v = outc[((size_t)b * 64 + (c - 64)) * 64 + hw];
  else              v = h[((size_t)b * 64 + (c - 128)) * 64 + hw] *
                        (notdone[b] ? 1.0f : 0.0f);
  in192[idx] = (_Float16)v;
}

// ---------------------------------------------------------------------------
// Pack conv weights [Cout][Cin][3][3] (f32) into WMMA B-fragments (f16):
// wpack[kc][ntile][lane][j], lane l holds column n=l&15, K rows 16*(l>>4)+j.
// GEMM K index k = kpos*Cin + ci  (kpos = ky*3+kx).
__global__ void prepack_kernel(const float* __restrict__ w,
                               _Float16* __restrict__ wpack,
                               int Cin, int Cout, int Kchunks) {
  int idx = blockIdx.x * 256 + threadIdx.x;
  int Ntiles = Cout >> 4;
  int total = Kchunks * Ntiles * 32;
  if (idx >= total) return;
  int lane = idx & 31;
  int t = idx >> 5;
  int ntile = t % Ntiles;
  int kc = t / Ntiles;
  int hi = lane >> 4, lm = lane & 15;
  int co = ntile * 16 + lm;
  _Float16* dst = wpack + (size_t)idx * 16;
#pragma unroll
  for (int j = 0; j < 16; ++j) {
    int k = kc * 32 + hi * 16 + j;
    int kpos = k / Cin;
    int ci = k - kpos * Cin;
    dst[j] = (_Float16)w[((size_t)co * Cin + ci) * 9 + kpos];
  }
}

// ---------------------------------------------------------------------------
// Implicit-GEMM 3x3 SAME conv on the WMMA pipe, register-blocked.
// Each wave computes a 64x32 output macro-tile: MT=4 M-tiles (one full batch
// sample, since HW==64) x NT=2 N-tiles, 8 independent f32 accumulators.
// Per 32-wide K chunk: 4 A-fragment loads + 2 B-fragment loads + 8 WMMAs.
// in:   f16 NHWC [B*HW][Cs] (first Cin channels used)
// gout: f32 NHWC [B*HW][Cout], bias added.
// grid = (BATCH, Cout/16/NT), block = 32 (one wave).
__global__ __launch_bounds__(32)
void conv_wmma_kernel(const _Float16* __restrict__ in, int Cs, int Cin,
                      const _Float16* __restrict__ wpack,
                      const float* __restrict__ bias,
                      float* __restrict__ gout, int Cout, int Kchunks) {
  const int lane = threadIdx.x;
  const int hi = lane >> 4;
  const int lm = lane & 15;
  const int bt = blockIdx.x;            // batch sample == 64-row M group
  const int n0t = blockIdx.y * NT;      // first N tile
  const int Ntiles = gridDim.y * NT;
  const int cpk = Cin >> 5;             // 32-wide K chunks per kernel tap

  int ybase[MT], xbase[MT];
#pragma unroll
  for (int u = 0; u < MT; ++u) {
    int pix = u * 16 + lm;              // pixel index within sample
    ybase[u] = pix >> 3;
    xbase[u] = pix & 7;
  }

  v8f acc[MT][NT] = {};
  for (int kc = 0; kc < Kchunks; ++kc) {
    const int kpos = kc / cpk;
    const int ci0 = (kc - kpos * cpk) << 5;
    const int ky = kpos / 3 - 1;
    const int kx = kpos % 3 - 1;

    v16h bf[NT];
#pragma unroll
    for (int nn = 0; nn < NT; ++nn)
      bf[nn] = *(const v16h*)(wpack +
                 (((size_t)kc * Ntiles + (n0t + nn)) * 32 + lane) * 16);

#pragma unroll
    for (int u = 0; u < MT; ++u) {
      const int yy = ybase[u] + ky, xx = xbase[u] + kx;
      v16h a;
      if (yy >= 0 && yy < 8 && xx >= 0 && xx < 8) {
        const _Float16* src =
            in + ((size_t)((bt << 6) + (yy << 3) + xx)) * Cs + ci0 + (hi << 4);
        const v8h lo = *(const v8h*)src;         // K = kc*32 + hi*16 + [0..7]
        const v8h h2 = *(const v8h*)(src + 8);   // K = kc*32 + hi*16 + 8 + [0..7]
#pragma unroll
        for (int j = 0; j < 8; ++j) { a[j] = lo[j]; a[8 + j] = h2[j]; }
      } else {
#pragma unroll
        for (int j = 0; j < 16; ++j) a[j] = (_Float16)0;
      }
#pragma unroll
      for (int nn = 0; nn < NT; ++nn)
        acc[u][nn] = __builtin_amdgcn_wmma_f32_16x16x32_f16(
            false, a, false, bf[nn], (short)0, acc[u][nn], false, false);
    }
  }

#pragma unroll
  for (int u = 0; u < MT; ++u) {
    const int mrow0 = (bt << 6) + u * 16 + (hi << 3);  // D: VGPR r -> M = r+8*hi
#pragma unroll
    for (int nn = 0; nn < NT; ++nn) {
      const int nbase = (n0t + nn) * 16 + lm;
      const float bc = bias[nbase];
#pragma unroll
      for (int r = 0; r < 8; ++r)
        gout[(size_t)(mrow0 + r) * Cout + nbase] = acc[u][nn][r] + bc;
    }
  }
}

// ---------------------------------------------------------------------------
// LSTM gate pointwise: gates f32 NHWC [b][hw][320] -> c1, o, a (NCHW f32)
__global__ void gates_kernel(const float* __restrict__ gates,
                             const float* __restrict__ c0n,
                             const unsigned char* __restrict__ notdone,
                             float* __restrict__ c1,
                             float* __restrict__ ogate,
                             float* __restrict__ agate) {
  int idx = blockIdx.x * 256 + threadIdx.x;
  if (idx >= BATCH * EDIM * HW) return;
  int b = idx >> 12;
  int r = idx & 4095;
  int e = r >> 6, hw = r & 63;
  const float* row = gates + ((size_t)b * 64 + hw) * 320;
  float gi = 1.0f / (1.0f + __expf(-row[e]));
  float gf = 1.0f / (1.0f + __expf(-row[64 + e]));
  float go = 1.0f / (1.0f + __expf(-row[128 + e]));
  float gg = tanhf(row[192 + e]);
  float ga = 1.0f / (1.0f + __expf(-row[256 + e]));
  size_t nchw = ((size_t)b * 64 + e) * 64 + hw;
  float cc = c0n[nchw] * (notdone[b] ? 1.0f : 0.0f);
  c1[nchw] = gf * cc + gi * gg;
  ogate[nchw] = go;
  agate[nchw] = ga;
}

// ---------------------------------------------------------------------------
// Attention over the rolling memory: one block per (b, head).
// kqv f32 NHWC [b][hw][192]; head h channels: k=h*24+[0,8), q=+[8,16), v=+[16,24)
// THD index d = hd*64 + hw.
__global__ __launch_bounds__(256)
void attn_kernel(const float* __restrict__ kqv,
                 const float* __restrict__ k0p,   // [B][NH][MEM][THD] (layer slice)
                 const float* __restrict__ v0p,
                 const float* __restrict__ posw,  // [MEM][NH*THD]
                 const float* __restrict__ posb,  // [MEM][NH]
                 const float* __restrict__ bias,  // [B][MEM]
                 _Float16* __restrict__ attn_out) // f16 NHWC [b][hw][64]
{
  const int b = blockIdx.x, nh = blockIdx.y;
  const int t = threadIdx.x;
  __shared__ float q_s[THD];
  __shared__ float kn_s[THD];
  __shared__ float sc_s[MEMN];
  const float invs = 0.044194173824159216f;  // 1/sqrt(512)

  for (int d = t; d < THD; d += 256) {
    int hd = d >> 6, hw = d & 63;
    const float* base = kqv + ((size_t)(b * 64 + hw)) * 192 + nh * 24;
    kn_s[d] = base[hd];
    q_s[d] = base[8 + hd] * invs;
  }
  __syncthreads();

  const int wave = t >> 5, lane = t & 31;
  const float* kmem = k0p + ((size_t)b * NHEAD + nh) * MEMN * THD;
  for (int m = wave; m < MEMN; m += 8) {
    float p = 0.0f;
    const float* pwm = posw + (size_t)m * (NHEAD * THD) + nh * THD;
    if (m < MEMN - 1) {
      const float* krow = kmem + (size_t)(m + 1) * THD;
      if (m + 9 < MEMN) __builtin_prefetch(kmem + (size_t)(m + 9) * THD, 0, 1);
      for (int d = lane; d < THD; d += 32) p += q_s[d] * (krow[d] + pwm[d]);
    } else {
      for (int d = lane; d < THD; d += 32) p += q_s[d] * (kn_s[d] + pwm[d]);
    }
#pragma unroll
    for (int off = 16; off > 0; off >>= 1) p += __shfl_xor(p, off, 32);
    if (lane == 0) sc_s[m] = p + bias[b * MEMN + m] + posb[m * NHEAD + nh];
  }
  __syncthreads();

  if (t == 0) {  // softmax normalize in place (40 elements)
    float mx = -1e30f;
    for (int m = 0; m < MEMN; ++m) mx = fmaxf(mx, sc_s[m]);
    float den = 0.0f;
    for (int m = 0; m < MEMN; ++m) { sc_s[m] = __expf(sc_s[m] - mx); den += sc_s[m]; }
    float rd = 1.0f / den;
    for (int m = 0; m < MEMN; ++m) sc_s[m] *= rd;
  }
  __syncthreads();

  const float* vmem = v0p + ((size_t)b * NHEAD + nh) * MEMN * THD;
  for (int d = t; d < THD; d += 256) {
    int hd = d >> 6, hw = d & 63;
    float acc = 0.0f;
    for (int m = 0; m < MEMN - 1; ++m)
      acc += sc_s[m] * vmem[(size_t)(m + 1) * THD + d];
    float vnew = kqv[((size_t)(b * 64 + hw)) * 192 + nh * 24 + 16 + hd];
    acc += sc_s[MEMN - 1] * vnew;
    attn_out[((size_t)b * 64 + hw) * 64 + nh * 8 + hd] = (_Float16)acc;
  }
}

// ---------------------------------------------------------------------------
// Residual + LayerNorm + cell update + hidden output. One block per sample b.
__global__ __launch_bounds__(256)
void final_kernel(const float* __restrict__ ao,   // f32 NHWC [b][hw][64]
                  const float* __restrict__ x,    // residual, NCHW
                  const float* __restrict__ lnw,  // [64][64] (E,HW)
                  const float* __restrict__ lnb,
                  const float* __restrict__ c1,
                  const float* __restrict__ og,
                  const float* __restrict__ ag,
                  float* __restrict__ outp)       // NCHW f32
{
  const int b = blockIdx.x, t = threadIdx.x;
  const int wave = t >> 5, lane = t & 31;
  __shared__ float red[16];
  float s = 0.0f, s2 = 0.0f;
  for (int i = t; i < EDIM * HW; i += 256) {
    int hw = i >> 6, e = i & 63;
    float v = ao[((size_t)b * 64 + hw) * 64 + e] +
              x[((size_t)b * 64 + e) * 64 + hw];
    s += v; s2 += v * v;
  }
#pragma unroll
  for (int off = 16; off > 0; off >>= 1) {
    s += __shfl_xor(s, off, 32);
    s2 += __shfl_xor(s2, off, 32);
  }
  if (lane == 0) { red[wave] = s; red[8 + wave] = s2; }
  __syncthreads();
  if (t == 0) {
    float ts = 0.0f, ts2 = 0.0f;
    for (int w = 0; w < 8; ++w) { ts += red[w]; ts2 += red[8 + w]; }
    red[0] = ts * (1.0f / 4096.0f);
    red[1] = ts2 * (1.0f / 4096.0f);
  }
  __syncthreads();
  const float mu = red[0];
  const float var = red[1] - mu * mu;
  const float rstd = rsqrtf(var + 1e-5f);
  for (int i = t; i < EDIM * HW; i += 256) {
    int hw = i >> 6, e = i & 63;
    size_t nchw = ((size_t)b * 64 + e) * 64 + hw;
    float v = ao[((size_t)b * 64 + hw) * 64 + e] + x[nchw];
    float aon = (v - mu) * rstd * lnw[e * 64 + hw] + lnb[e * 64 + hw];
    float cn = c1[nchw] + ag[nchw] * tanhf(aon);
    outp[nchw] = og[nchw] * tanhf(cn);
  }
}

// ---------------------------------------------------------------------------
extern "C" void kernel_launch(void* const* d_in, const int* in_sizes, int n_in,
                              void* d_out, int out_size, void* d_ws,
                              size_t ws_size, hipStream_t stream) {
  const float* x       = (const float*)d_in[0];
  const float* h0      = (const float*)d_in[1];
  const float* c0      = (const float*)d_in[2];
  const float* k0      = (const float*)d_in[3];
  const float* v0      = (const float*)d_in[4];
  const unsigned char* src_mask = (const unsigned char*)d_in[5];
  const unsigned char* notdone  = (const unsigned char*)d_in[6];
  const float* main_w  = (const float*)d_in[7];
  const float* main_b  = (const float*)d_in[8];
  const float* proj_w  = (const float*)d_in[9];
  const float* proj_b  = (const float*)d_in[10];
  const float* out_w   = (const float*)d_in[11];
  const float* out_b   = (const float*)d_in[12];
  const float* ln_w    = (const float*)d_in[13];
  const float* ln_b    = (const float*)d_in[14];
  const float* pos_w   = (const float*)d_in[15];
  const float* pos_b   = (const float*)d_in[16];

  // bump allocator over workspace
  char* ws = (char*)d_ws;
  size_t off = 0;
  auto carve = [&](size_t bytes) -> char* {
    char* p = ws + off;
    off += (bytes + 255) & ~(size_t)255;
    return p;
  };
  float*    bias_buf = (float*)carve(BATCH * MEMN * 4);
  float*    out_cur  = (float*)carve(BATCH * EDIM * HW * 4);
  _Float16* in192    = (_Float16*)carve((size_t)BATCH * HW * 192 * 2);
  _Float16* wp_main  = (_Float16*)carve((size_t)54 * 20 * 32 * 16 * 2);
  _Float16* wp_proj  = (_Float16*)carve((size_t)36 * 12 * 32 * 16 * 2);
  _Float16* wp_out   = (_Float16*)carve((size_t)18 * 4 * 32 * 16 * 2);
  float*    gates    = (float*)carve((size_t)BATCH * HW * 320 * 4);
  float*    kqv      = (float*)carve((size_t)BATCH * HW * 192 * 4);
  float*    c1b      = (float*)carve(BATCH * EDIM * HW * 4);
  float*    ogb      = (float*)carve(BATCH * EDIM * HW * 4);
  float*    agb      = (float*)carve(BATCH * EDIM * HW * 4);
  _Float16* attn_h   = (_Float16*)carve((size_t)BATCH * HW * 64 * 2);
  float*    ao       = (float*)carve((size_t)BATCH * HW * 64 * 4);

  const size_t hc_stride = (size_t)BATCH * EDIM * HW;          // per-layer h0/c0
  const size_t kv_stride = (size_t)BATCH * NHEAD * MEMN * THD; // per-layer k0/v0

  bias_kernel<<<(BATCH * MEMN + 255) / 256, 256, 0, stream>>>(src_mask, notdone,
                                                              bias_buf);
  init_out_kernel<<<(int)(hc_stride / 256), 256, 0, stream>>>(
      h0 + (LAYERS - 1) * hc_stride, notdone, out_cur);

  for (int n = 0; n < LAYERS; ++n) {
    build_input_kernel<<<(BATCH * HW * 192) / 256, 256, 0, stream>>>(
        x, out_cur, h0 + n * hc_stride, notdone, in192);

    // main conv: Cin=192, Cout=320, K=1728 (54 chunks), Ntiles=20
    prepack_kernel<<<(54 * 20 * 32 + 255) / 256, 256, 0, stream>>>(
        main_w + (size_t)n * 320 * 192 * 9, wp_main, 192, 320, 54);
    conv_wmma_kernel<<<dim3(BATCH, 20 / NT), 32, 0, stream>>>(
        in192, 192, 192, wp_main, main_b + n * 320, gates, 320, 54);

    gates_kernel<<<(int)(hc_stride / 256), 256, 0, stream>>>(
        gates, c0 + n * hc_stride, notdone, c1b, ogb, agb);

    // proj conv: Cin=128 (first 128 chans of in192), Cout=192, K=1152, Ntiles=12
    prepack_kernel<<<(36 * 12 * 32 + 255) / 256, 256, 0, stream>>>(
        proj_w + (size_t)n * 192 * 128 * 9, wp_proj, 128, 192, 36);
    conv_wmma_kernel<<<dim3(BATCH, 12 / NT), 32, 0, stream>>>(
        in192, 192, 128, wp_proj, proj_b + n * 192, kqv, 192, 36);

    attn_kernel<<<dim3(BATCH, NHEAD), 256, 0, stream>>>(
        kqv, k0 + n * kv_stride, v0 + n * kv_stride,
        pos_w + (size_t)n * MEMN * NHEAD * THD, pos_b + n * MEMN * NHEAD,
        bias_buf, attn_h);

    // out conv: Cin=64, Cout=64, K=576 (18 chunks), Ntiles=4
    prepack_kernel<<<(18 * 4 * 32 + 255) / 256, 256, 0, stream>>>(
        out_w + (size_t)n * 64 * 64 * 9, wp_out, 64, 64, 18);
    conv_wmma_kernel<<<dim3(BATCH, 4 / NT), 32, 0, stream>>>(
        attn_h, 64, 64, wp_out, out_b + n * 64, ao, 64, 18);

    float* dst = (n == LAYERS - 1) ? (float*)d_out : out_cur;
    final_kernel<<<BATCH, 256, 0, stream>>>(
        ao, x, ln_w + (size_t)n * EDIM * HW, ln_b + (size_t)n * EDIM * HW,
        c1b, ogb, agb, dst);
  }
}
